// GatedCrossAttention_53145925321335
// MI455X (gfx1250) — matrix operand order repaired and checked by reading, not compiled, not run on hardware
//
#include <hip/hip_runtime.h>
#include <stdint.h>

// ---------------------------------------------------------------------------
// Types for CDNA5 WMMA (wave32): A/B = 16 x bf16 per lane, C/D = 8 x f32.
// ---------------------------------------------------------------------------
typedef __attribute__((ext_vector_type(16))) __bf16 v16bf;
typedef __attribute__((ext_vector_type(8)))  __bf16 v8bf;
typedef __attribute__((ext_vector_type(8)))  float  v8f;
typedef __attribute__((ext_vector_type(4)))  float  v4f;

__device__ __forceinline__ __bf16 f2bf(float f) {
  uint32_t u = __builtin_bit_cast(uint32_t, f);
  uint32_t r = u + 0x7FFFu + ((u >> 16) & 1u);   // round-to-nearest-even
  uint16_t h = (uint16_t)(r >> 16);
  return __builtin_bit_cast(__bf16, h);
}
__device__ __forceinline__ uint32_t f2bf_pack(float lo, float hi) {
  uint16_t l = __builtin_bit_cast(uint16_t, f2bf(lo));
  uint16_t h = __builtin_bit_cast(uint16_t, f2bf(hi));
  return (uint32_t)l | ((uint32_t)h << 16);
}
__device__ __forceinline__ float bf2f(__bf16 x) {
  uint16_t h = __builtin_bit_cast(uint16_t, x);
  uint32_t u = ((uint32_t)h) << 16;
  return __builtin_bit_cast(float, u);
}
__device__ __forceinline__ float sigmoidf_(float x) { return 1.0f / (1.0f + __expf(-x)); }
__device__ __forceinline__ float siluf_(float x)    { return x * sigmoidf_(x); }

__device__ __forceinline__ v16bf loadA16(const __bf16* p) {
  v8bf lo = *(const v8bf*)(p);
  v8bf hi = *(const v8bf*)(p + 8);
  return __builtin_shufflevector(lo, hi, 0,1,2,3,4,5,6,7,8,9,10,11,12,13,14,15);
}

#define WMMA_BF16(A_, B_, C_) \
  __builtin_amdgcn_wmma_f32_16x16x32_bf16(false, (A_), false, (B_), (short)0, (C_), false, false)

// ---------------------------------------------------------------------------
// Core WMMA routine: one wave computes a 32(M) x 64(N) f32 tile of
//   C = A[M,K] (row-major bf16)  @  B[N,K]^T (row-major-[N,K] bf16)
// K-loop in steps of 32 via v_wmma_f32_16x16x32_bf16; manually double-
// buffered: iteration k issues loads for k+1 (fresh regs) before its 8
// WMMAs, so waits only cover the in-flight next-iteration loads.
// Six row pointers are hoisted and stay live; in-loop addressing is pure
// immediate offsets. Kernels use __launch_bounds__(256, 2) so the VGPR
// budget (2 waves/SIMD) comfortably holds both fragment buffer sets ->
// no spills, no pointer remat, no WMMA->VALU hazard NOPs.
//
// A fragment (same contiguous K order as B): lane L<16 holds row M=L,
// K = kb..kb+15; lanes 16..31 hold K = kb+16..kb+31.
// B fragment ([N,K] storage is K-contiguous per N-row): lane holds
// N = L&15, K = kb + (L>=16 ? 16 : 0) + j  -> one 32-byte load per lane.
// ---------------------------------------------------------------------------
__device__ __forceinline__ void wmma_tile32x64(
    const __bf16* __restrict__ A, const __bf16* __restrict__ Bm,
    int K, int lda, int ldb, int mBase, int nBase, v8f acc[2][4])
{
  const int lane = threadIdx.x & 31;
  const int l16  = lane & 15;
  const int hi   = lane >> 4;
  const __bf16* aP0 = A + (size_t)(mBase + l16) * (size_t)lda + (hi ? 16 : 0);
  const __bf16* aP1 = aP0 + (size_t)16 * (size_t)lda;
  const __bf16* bP0 = Bm + (size_t)(nBase + l16) * (size_t)ldb + (hi ? 16 : 0);
  const __bf16* bP1 = bP0 + (size_t)16 * (size_t)ldb;
  const __bf16* bP2 = bP0 + (size_t)32 * (size_t)ldb;
  const __bf16* bP3 = bP0 + (size_t)48 * (size_t)ldb;

  // Prologue: fragments for kb = 0
  v16bf af0 = loadA16(aP0);
  v16bf af1 = loadA16(aP1);
  v16bf b0  = *(const v16bf*)(bP0);
  v16bf b1  = *(const v16bf*)(bP1);
  v16bf b2  = *(const v16bf*)(bP2);
  v16bf b3  = *(const v16bf*)(bP3);

#pragma unroll 2
  for (int kb = 32; kb < K; kb += 32) {
    // Prefetch the A stream ahead (folds to immediate offsets of live ptrs)
    __builtin_prefetch(aP0 + kb + 256, 0, 3);     // global_prefetch_b8
    __builtin_prefetch(aP1 + kb + 256, 0, 3);
    // Issue next-iteration loads first (fresh registers; no WAR with WMMAs)
    v16bf naf0 = loadA16(aP0 + kb);
    v16bf naf1 = loadA16(aP1 + kb);
    v16bf nb0  = *(const v16bf*)(bP0 + kb);
    v16bf nb1  = *(const v16bf*)(bP1 + kb);
    v16bf nb2  = *(const v16bf*)(bP2 + kb);
    v16bf nb3  = *(const v16bf*)(bP3 + kb);
    // 8 WMMAs on the previous iteration's fragments
    acc[0][0] = WMMA_BF16(af0, b0, acc[0][0]);
    acc[1][0] = WMMA_BF16(af1, b0, acc[1][0]);
    acc[0][1] = WMMA_BF16(af0, b1, acc[0][1]);
    acc[1][1] = WMMA_BF16(af1, b1, acc[1][1]);
    acc[0][2] = WMMA_BF16(af0, b2, acc[0][2]);
    acc[1][2] = WMMA_BF16(af1, b2, acc[1][2]);
    acc[0][3] = WMMA_BF16(af0, b3, acc[0][3]);
    acc[1][3] = WMMA_BF16(af1, b3, acc[1][3]);
    // Buffer swap (register renaming under unroll-by-2)
    af0 = naf0; af1 = naf1; b0 = nb0; b1 = nb1; b2 = nb2; b3 = nb3;
  }
  // Epilogue: last K-step
  acc[0][0] = WMMA_BF16(af0, b0, acc[0][0]);
  acc[1][0] = WMMA_BF16(af1, b0, acc[1][0]);
  acc[0][1] = WMMA_BF16(af0, b1, acc[0][1]);
  acc[1][1] = WMMA_BF16(af1, b1, acc[1][1]);
  acc[0][2] = WMMA_BF16(af0, b2, acc[0][2]);
  acc[1][2] = WMMA_BF16(af1, b2, acc[1][2]);
  acc[0][3] = WMMA_BF16(af0, b3, acc[0][3]);
  acc[1][3] = WMMA_BF16(af1, b3, acc[1][3]);
}

// C/D layout: VGPR i -> M = mBase + mt*16 + (laneHi?8:0) + i,
//             N = nBase + t*16 + (lane&15)
// Block = 256 thr = 8 waves = 4(M) x 2(N) waves -> block tile 128 x 128.
#define GEMM_PROLOGUE()                                            \
  const int wave  = threadIdx.x >> 5;                              \
  const int mBase = blockIdx.y * 128 + (wave & 3) * 32;            \
  const int nBase = blockIdx.x * 128 + (wave >> 2) * 64;           \
  v8f z = {0,0,0,0,0,0,0,0};                                       \
  v8f acc[2][4] = {{z,z,z,z},{z,z,z,z}};                           \
  const int lane = threadIdx.x & 31, l16 = lane & 15, hi = lane >> 4;

// ---------------------------------------------------------------------------
// Phase-0 kernels
// ---------------------------------------------------------------------------
__global__ void k_layernorm_bf16(const float* __restrict__ x,
                                 const float* __restrict__ g,
                                 const float* __restrict__ b,
                                 __bf16* __restrict__ out, int E)
{
  __shared__ float s1[256], s2[256];
  const int row = blockIdx.x;
  const float* xr = x + (size_t)row * E;
  float sum = 0.f, sq = 0.f;
  for (int i = threadIdx.x; i < E; i += 256) { float v = xr[i]; sum += v; sq += v * v; }
  s1[threadIdx.x] = sum; s2[threadIdx.x] = sq;
  __syncthreads();
  for (int off = 128; off > 0; off >>= 1) {
    if ((int)threadIdx.x < off) {
      s1[threadIdx.x] += s1[threadIdx.x + off];
      s2[threadIdx.x] += s2[threadIdx.x + off];
    }
    __syncthreads();
  }
  const float mu  = s1[0] / (float)E;
  const float var = s2[0] / (float)E - mu * mu;
  const float rs  = rsqrtf(var + 1e-5f);
  __bf16* orow = out + (size_t)row * E;
  for (int i = threadIdx.x; i < E; i += 256)
    orow[i] = f2bf((xr[i] - mu) * rs * g[i] + b[i]);
}

// Vectorized f32 -> bf16 conversion (n must be a multiple of 4; all sizes are)
__global__ void k_f32_to_bf16(const float* __restrict__ in,
                              uint32_t* __restrict__ out, size_t n4)
{
  size_t i = (size_t)blockIdx.x * blockDim.x + threadIdx.x;
  size_t stride = (size_t)gridDim.x * blockDim.x;
  for (; i < n4; i += stride) {
    v4f v = *(const v4f*)(in + i * 4);
    out[i * 2]     = f2bf_pack(v[0], v[1]);
    out[i * 2 + 1] = f2bf_pack(v[2], v[3]);
  }
}

__global__ void k_invlen(const unsigned char* __restrict__ mask,
                         float* __restrict__ invLen, int C)
{
  __shared__ float s[256];
  const int b = blockIdx.x;
  float acc = 0.f;
  for (int i = threadIdx.x; i < C; i += 256)
    acc += mask[(size_t)b * C + i] ? 0.f : 1.f;
  s[threadIdx.x] = acc; __syncthreads();
  for (int off = 128; off > 0; off >>= 1) {
    if ((int)threadIdx.x < off) s[threadIdx.x] += s[threadIdx.x + off];
    __syncthreads();
  }
  if (threadIdx.x == 0) invLen[b] = s[0] > 0.f ? 1.f / s[0] : 0.f;
}

// ---------------------------------------------------------------------------
// GEMM kernels (8 wave32 waves/block; 2 waves/SIMD -> big VGPR budget)
// ---------------------------------------------------------------------------
__global__ void __launch_bounds__(256, 2)
k_gemm_base(const __bf16* __restrict__ qn, const __bf16* __restrict__ Wq,
            const float* __restrict__ bq,
            __bf16* __restrict__ u, __bf16* __restrict__ r,
            __bf16* __restrict__ q, int K, int E, int Z)
{
  GEMM_PROLOGUE();
  wmma_tile32x64(qn, Wq, K, K, K, mBase, nBase, acc);
#pragma unroll
  for (int t = 0; t < 4; ++t) {
    const int n = nBase + t * 16 + l16;
    const float bias = bq[n];
#pragma unroll
    for (int mt = 0; mt < 2; ++mt) {
#pragma unroll
      for (int i = 0; i < 8; ++i) {
        const int m = mBase + mt * 16 + hi * 8 + i;
        float x = acc[mt][t][i] + bias;
        if (n < E)          u[(size_t)m * E + n]           = f2bf(sigmoidf_(x));
        else if (n < 2 * E) r[(size_t)m * E + (n - E)]     = f2bf(siluf_(x));
        else                q[(size_t)m * Z + (n - 2 * E)] = f2bf(x);
      }
    }
  }
}

__global__ void __launch_bounds__(256, 2)
k_gemm_k(const __bf16* __restrict__ keyb, const __bf16* __restrict__ Wk,
         const float* __restrict__ bk, __bf16* __restrict__ kout,
         int K, int Z)
{
  GEMM_PROLOGUE();
  wmma_tile32x64(keyb, Wk, K, K, K, mBase, nBase, acc);
#pragma unroll
  for (int t = 0; t < 4; ++t) {
    const int n = nBase + t * 16 + l16;
    const float bias = bk[n];
#pragma unroll
    for (int mt = 0; mt < 2; ++mt) {
#pragma unroll
      for (int i = 0; i < 8; ++i) {
        const int m = mBase + mt * 16 + hi * 8 + i;
        kout[(size_t)m * Z + n] = f2bf(acc[mt][t][i] + bias);
      }
    }
  }
}

__global__ void __launch_bounds__(256, 2)
k_gemm_v(const __bf16* __restrict__ keyb, const __bf16* __restrict__ Wv,
         const float* __restrict__ bv, __bf16* __restrict__ vT,
         int K, int E, int C)
{
  GEMM_PROLOGUE();
  wmma_tile32x64(keyb, Wv, K, K, K, mBase, nBase, acc);
#pragma unroll
  for (int t = 0; t < 4; ++t) {
    const int n = nBase + t * 16 + l16;
    const float bias = bv[n];
#pragma unroll
    for (int mt = 0; mt < 2; ++mt) {
#pragma unroll
      for (int i = 0; i < 8; ++i) {
        const int m = mBase + mt * 16 + hi * 8 + i;   // m over B*C
        const int b = m / C, c = m % C;
        // store transposed: vT[b][n][c]  (K-major for the h GEMM)
        vT[((size_t)b * E + n) * (size_t)C + c] = f2bf(siluf_(acc[mt][t][i] + bias));
      }
    }
  }
}

__global__ void __launch_bounds__(256, 2)
k_gemm_attn(const __bf16* __restrict__ qb, const __bf16* __restrict__ kb,
            const float* __restrict__ rel,
            const unsigned char* __restrict__ mask,
            const float* __restrict__ invLen,
            float* __restrict__ attnF, __bf16* __restrict__ attnB,
            int Z, int S, int C, int maxp)
{
  const int b = blockIdx.z;
  const __bf16* A  = qb + (size_t)b * S * Z;
  const __bf16* Bm = kb + (size_t)b * C * Z;
  GEMM_PROLOGUE();
  wmma_tile32x64(A, Bm, Z, Z, Z, mBase, nBase, acc);
  const float il = invLen[b];
#pragma unroll
  for (int t = 0; t < 4; ++t) {
    const int c = nBase + t * 16 + l16;
    const float im = mask[(size_t)b * C + c] ? 0.f : 1.f;
#pragma unroll
    for (int mt = 0; mt < 2; ++mt) {
#pragma unroll
      for (int i = 0; i < 8; ++i) {
        const int s = mBase + mt * 16 + hi * 8 + i;
        float x = acc[mt][t][i] * il + rel[maxp - 1 + c - s];
        x = fmaxf(x, 0.f);
        x = x * x * im;
        const size_t idx = ((size_t)b * S + s) * (size_t)C + c;
        attnF[idx] = x;
        attnB[idx] = f2bf(x);
      }
    }
  }
}

__global__ void __launch_bounds__(256, 2)
k_gemm_h(const __bf16* __restrict__ attnB, const __bf16* __restrict__ vT,
         const __bf16* __restrict__ r, __bf16* __restrict__ hr,
         int C, int E, int S)
{
  const int b = blockIdx.z;
  const __bf16* A  = attnB + (size_t)b * S * C;
  const __bf16* Bm = vT    + (size_t)b * E * C;
  GEMM_PROLOGUE();
  wmma_tile32x64(A, Bm, C, C, C, mBase, nBase, acc);
#pragma unroll
  for (int t = 0; t < 4; ++t) {
    const int n = nBase + t * 16 + l16;
#pragma unroll
    for (int mt = 0; mt < 2; ++mt) {
#pragma unroll
      for (int i = 0; i < 8; ++i) {
        const int s = mBase + mt * 16 + hi * 8 + i;
        const size_t g = (size_t)b * S + s;
        hr[g * E + n] = f2bf(acc[mt][t][i] * bf2f(r[g * E + n]));  // fuse h * r
      }
    }
  }
}

__global__ void __launch_bounds__(256, 2)
k_gemm_final(const __bf16* __restrict__ hr, const __bf16* __restrict__ Wh,
             const float* __restrict__ bh,
             const float* __restrict__ query, const __bf16* __restrict__ u,
             float* __restrict__ out, int K, int E)
{
  GEMM_PROLOGUE();
  wmma_tile32x64(hr, Wh, K, K, K, mBase, nBase, acc);
#pragma unroll
  for (int t = 0; t < 4; ++t) {
    const int n = nBase + t * 16 + l16;
    const float bias = bh[n];
#pragma unroll
    for (int mt = 0; mt < 2; ++mt) {
#pragma unroll
      for (int i = 0; i < 8; ++i) {
        const int m = mBase + mt * 16 + hi * 8 + i;
        const size_t idx = (size_t)m * E + n;
        const float hh = siluf_(acc[mt][t][i] + bias);
        const float qv = query[idx];
        out[idx] = qv + bf2f(u[idx]) * (hh - qv);
      }
    }
  }
}

// ---------------------------------------------------------------------------
// Host launcher
// ---------------------------------------------------------------------------
extern "C" void kernel_launch(void* const* d_in, const int* in_sizes, int n_in,
                              void* d_out, int out_size, void* d_ws, size_t ws_size,
                              hipStream_t stream)
{
  (void)n_in; (void)out_size; (void)ws_size;
  const int B = 8, S = 2048, C = 2048, E = 1024, Z = 128;
  const int EZ2 = 2 * E + Z;                       // 2176
  const int maxp = (in_sizes[14] + 1) / 2;         // 2048

  const float* query = (const float*)d_in[0];
  const float* key   = (const float*)d_in[1];
  /* d_in[2] (value) is unused by the reference */
  const unsigned char* mask = (const unsigned char*)d_in[3];
  const float* ln_g = (const float*)d_in[4];
  const float* ln_b = (const float*)d_in[5];
  const float* Wq = (const float*)d_in[6];
  const float* bq = (const float*)d_in[7];
  const float* Wk = (const float*)d_in[8];
  const float* bk = (const float*)d_in[9];
  const float* Wv = (const float*)d_in[10];
  const float* bv = (const float*)d_in[11];
  const float* Wh = (const float*)d_in[12];
  const float* bh = (const float*)d_in[13];
  const float* rel = (const float*)d_in[14];

  char* ws = (char*)d_ws;
  size_t off = 0;
  auto alloc = [&](size_t bytes) -> char* {
    char* p = ws + off;
    off += (bytes + 255) & ~(size_t)255;
    return p;
  };
  __bf16* qn    = (__bf16*)alloc((size_t)B * S * E * 2);   // dead after k_gemm_base
  __bf16* keyb  = (__bf16*)alloc((size_t)B * C * E * 2);   // dead after k_gemm_v
  __bf16* attnB = (__bf16*)d_ws;   // 67 MB; aliases qn+keyb which are dead by then
  __bf16* Wqb = (__bf16*)alloc((size_t)EZ2 * E * 2);
  __bf16* ub  = (__bf16*)alloc((size_t)B * S * E * 2);
  __bf16* rb  = (__bf16*)alloc((size_t)B * S * E * 2);
  __bf16* qb  = (__bf16*)alloc((size_t)B * S * Z * 2);
  __bf16* Wkb = (__bf16*)alloc((size_t)Z * E * 2);
  __bf16* Wvb = (__bf16*)alloc((size_t)E * E * 2);
  __bf16* Whb = (__bf16*)alloc((size_t)E * E * 2);
  __bf16* kb  = (__bf16*)alloc((size_t)B * C * Z * 2);
  __bf16* vT  = (__bf16*)alloc((size_t)B * E * C * 2);
  __bf16* hr  = (__bf16*)alloc((size_t)B * S * E * 2);
  float*  invLen = (float*)alloc(256);

  float* outMain = (float*)d_out;                // (B,S,E)
  float* outAttn = outMain + (size_t)B * S * E;  // (B,S,C)

  const dim3 blk(256);

  // Phase 0: precision conversion + layernorm + lengths
  k_layernorm_bf16<<<B * S, 256, 0, stream>>>(query, ln_g, ln_b, qn, E);
  k_f32_to_bf16<<<1024, 256, 0, stream>>>(key, (uint32_t*)keyb, (size_t)B * C * E / 4);
  k_f32_to_bf16<<<256, 256, 0, stream>>>(Wq, (uint32_t*)Wqb, (size_t)EZ2 * E / 4);
  k_f32_to_bf16<<<32, 256, 0, stream>>>(Wk, (uint32_t*)Wkb, (size_t)Z * E / 4);
  k_f32_to_bf16<<<128, 256, 0, stream>>>(Wv, (uint32_t*)Wvb, (size_t)E * E / 4);
  k_f32_to_bf16<<<128, 256, 0, stream>>>(Wh, (uint32_t*)Whb, (size_t)E * E / 4);
  k_invlen<<<B, 256, 0, stream>>>(mask, invLen, C);

  // Phase 1: projections (block tile 128x128)
  k_gemm_base<<<dim3(EZ2 / 128, (B * S) / 128), blk, 0, stream>>>(qn, Wqb, bq, ub, rb, qb, E, E, Z);
  k_gemm_k<<<dim3(Z / 128, (B * C) / 128), blk, 0, stream>>>(keyb, Wkb, bk, kb, E, Z);
  k_gemm_v<<<dim3(E / 128, (B * C) / 128), blk, 0, stream>>>(keyb, Wvb, bv, vT, E, E, C);

  // Phase 2: attention scores (writes f32 attn output + bf16 staging)
  k_gemm_attn<<<dim3(C / 128, S / 128, B), blk, 0, stream>>>(qb, kb, rel, mask, invLen,
                                                             outAttn, attnB, Z, S, C, maxp);

  // Phase 3: h = attn @ v, fused with * r
  k_gemm_h<<<dim3(E / 128, S / 128, B), blk, 0, stream>>>(attnB, vT, rb, hr, C, E, S);

  // Phase 4: final projection, silu, gated residual
  k_gemm_final<<<dim3(E / 128, (B * S) / 128), blk, 0, stream>>>(hr, Whb, bh, query, ub,
                                                                 outMain, E, E);
}
